// HybridQLSTM_65481071409480
// MI455X (gfx1250) — compile-verified, matrix operand order checked by reading, not run on hardware
//
#include <hip/hip_runtime.h>
#include <hip/hip_bf16.h>

typedef __attribute__((ext_vector_type(16))) __bf16 bf16x16;
typedef __attribute__((ext_vector_type(8)))  __bf16 bf16x8;
typedef __attribute__((ext_vector_type(8)))  float  f32x8;
typedef __attribute__((ext_vector_type(4)))  float  f32x4;
typedef __attribute__((ext_vector_type(4)))  int    i32x4;
typedef __attribute__((address_space(1)))    i32x4  as1_i32x4;
typedef __attribute__((address_space(3)))    i32x4  as3_i32x4;

namespace {
constexpr int SEQ   = 512;
constexpr int BATCH = 32;
constexpr int EMB   = 512;
constexpr int HID   = 512;
constexpr int TAGS  = 50;
constexpr int G4    = 4 * HID;        // 2048 gate width
constexpr int ROWS  = SEQ * BATCH;    // 16384 rows of x / xg
constexpr int LDA   = 520;            // padded LDS row (bf16) -> bank shift 4/row
}

__device__ __forceinline__ bf16x16 join8(bf16x8 lo, bf16x8 hi) {
  bf16x16 r;
#pragma unroll
  for (int i = 0; i < 8; ++i) { r[i] = lo[i]; r[i + 8] = hi[i]; }
  return r;
}

// ---- embedding gather + f32 -> bf16 (x_bf [ROWS, EMB]) -------------------
__global__ void k_gather_cvt(const int* __restrict__ sent,
                             const float* __restrict__ emb,
                             __bf16* __restrict__ xbf) {
  int t   = blockIdx.x * blockDim.x + threadIdx.x;  // exact ROWS*EMB/8 threads
  int row = t >> 6;                                 // EMB/8 = 64 chunks/row
  int c8  = t & 63;
  int tok = sent[row];
  const float* s = emb + (size_t)tok * EMB + c8 * 8;
  f32x4 a = *(const f32x4*)s;
  f32x4 b = *(const f32x4*)(s + 4);
  bf16x8 o;
#pragma unroll
  for (int i = 0; i < 4; ++i) { o[i] = (__bf16)a[i]; o[i + 4] = (__bf16)b[i]; }
  *(bf16x8*)(xbf + (size_t)row * EMB + c8 * 8) = o;
}

// ---- f32 -> bf16 weight convert (exact n/8 threads) ----------------------
__global__ void k_cvt_bf16(const float* __restrict__ src, __bf16* __restrict__ dst) {
  int t = blockIdx.x * blockDim.x + threadIdx.x;
  const float* s = src + (size_t)t * 8;
  f32x4 a = *(const f32x4*)s;
  f32x4 b = *(const f32x4*)(s + 4);
  bf16x8 o;
#pragma unroll
  for (int i = 0; i < 4; ++i) { o[i] = (__bf16)a[i]; o[i + 4] = (__bf16)b[i]; }
  *(bf16x8*)(dst + (size_t)t * 8) = o;
}

// ---- bias = b_ih + b_hh (exact G4 threads) -------------------------------
__global__ void k_bias(const float* __restrict__ bi, const float* __restrict__ bh,
                       float* __restrict__ bias) {
  int t = blockIdx.x * blockDim.x + threadIdx.x;
  bias[t] = bi[t] + bh[t];
}

// ---- xg = x @ W_ih^T + bias : [ROWS, G4] WMMA bf16 GEMM ------------------
// Block = 8 waves sharing one 16-row A strip staged in LDS (async copy).
// Each wave computes a 16x64 strip (4 N-tiles).
__global__ __launch_bounds__(256) void k_gemm_xg(const __bf16* __restrict__ X,
                                                 const __bf16* __restrict__ W,
                                                 const float* __restrict__ bias,
                                                 float* __restrict__ XG) {
  __shared__ __align__(16) __bf16 ldsA[16 * LDA];   // ~16.3 KiB, padded rows

  const int lane = threadIdx.x & 31;
  const int wv   = threadIdx.x >> 5;
  const int m0   = (blockIdx.x >> 2) * 16;                 // shared by the block
  const int n0   = (((blockIdx.x & 3) * 8 + wv)) * 64;     // per-wave 64-col strip
  const int idx  = lane & 15;
  const int half = lane >> 4;

  // ---- stage A strip [16 x EMB] into LDS: 1024 16B chunks, 4 per thread ----
#pragma unroll
  for (int i = 0; i < 4; ++i) {
    const int q   = threadIdx.x + i * 256;
    const int row = q >> 6;
    const int c   = q & 63;
    const __bf16* g = X + (size_t)(m0 + row) * EMB + c * 8;
    __bf16*       l = &ldsA[row * LDA + c * 8];
#if __has_builtin(__builtin_amdgcn_global_load_async_to_lds_b128)
    __builtin_amdgcn_global_load_async_to_lds_b128(
        (as1_i32x4*)g, (as3_i32x4*)l, 0, 0);
#else
    *(bf16x8*)l = *(const bf16x8*)g;
#endif
  }
#if __has_builtin(__builtin_amdgcn_global_load_async_to_lds_b128)
#if __has_builtin(__builtin_amdgcn_s_wait_asynccnt)
  __builtin_amdgcn_s_wait_asynccnt(0);
#else
  asm volatile("s_wait_asynccnt 0x0" ::: "memory");
#endif
#endif
  __syncthreads();

  f32x8 acc[4];
#pragma unroll
  for (int t = 0; t < 4; ++t) {
    float bv = bias[n0 + t * 16 + idx];
#pragma unroll
    for (int r = 0; r < 8; ++r) acc[t][r] = bv;
  }

  const __bf16* aptr = &ldsA[idx * LDA];
#pragma unroll 1
  for (int kb = 0; kb < EMB / 32; ++kb) {
    const int ka = kb * 32 + half * 8;
    // A fragment from LDS (ds_load_b128 x2, conflict-free via row padding)
    bf16x16 A = join8(*(const bf16x8*)(aptr + ka),
                      *(const bf16x8*)(aptr + ka + 16));
    // Issue all four B tile load pairs before any WMMA (staggered waits)
    bf16x16 B[4];
#pragma unroll
    for (int t = 0; t < 4; ++t) {
      const __bf16* brow = W + (size_t)(n0 + t * 16 + idx) * EMB + ka;
      B[t] = join8(*(const bf16x8*)brow, *(const bf16x8*)(brow + 16));
      __builtin_prefetch((const void*)(brow + 32), 0, 1);
    }
#pragma unroll
    for (int t = 0; t < 4; ++t)
      acc[t] = __builtin_amdgcn_wmma_f32_16x16x32_bf16(
          false, A, false, B[t], (short)0, acc[t], false, false);
  }

#pragma unroll
  for (int t = 0; t < 4; ++t) {
    const int n = n0 + t * 16 + idx;
#pragma unroll
    for (int r = 0; r < 8; ++r)
      XG[(size_t)(m0 + half * 8 + r) * G4 + n] = acc[t][r];
  }
}

// ---- sequential LSTM on one WGP: 32 waves, h in LDS as bf16 --------------
__global__ __launch_bounds__(1024) void k_lstm_seq(const float* __restrict__ XG,
                                                   const __bf16* __restrict__ Whh,
                                                   float* __restrict__ gates,
                                                   float* __restrict__ cbuf,
                                                   float* __restrict__ hf,
                                                   float* __restrict__ hsum_out) {
  __shared__ __align__(16) __bf16 hbf[BATCH * HID];  // 32 KiB
  const int tid  = threadIdx.x;
  const int lane = tid & 31;
  const int wv   = tid >> 5;                          // 0..31
  const int idx  = lane & 15;
  const int half = lane >> 4;
  const int m0     = (wv & 1) * 16;                   // batch tile
  const int ntbase = (wv >> 1) * 8;                   // 8 N-tiles per wave

  for (int i = tid; i < BATCH * HID; i += 1024) {
    hbf[i]  = (__bf16)0.0f;
    cbuf[i] = 0.0f;
  }
  __syncthreads();

#pragma unroll 1
  for (int s = 0; s < SEQ; ++s) {
    const float* xg_s = XG + (size_t)s * BATCH * G4;

    // phase 1: gates = xg[s] + h @ Whh^T   (WMMA, C initialized from xg)
#pragma unroll 1
    for (int nt = 0; nt < 8; ++nt) {
      const int n = (ntbase + nt) * 16 + idx;
      f32x8 acc;
#pragma unroll
      for (int r = 0; r < 8; ++r)
        acc[r] = xg_s[(size_t)(m0 + half * 8 + r) * G4 + n];
      const __bf16* arow = hbf + (m0 + idx) * HID;
      const __bf16* brow = Whh + (size_t)n * HID;
#pragma unroll 1
      for (int kb = 0; kb < HID / 32; ++kb) {
        const int ka = kb * 32 + half * 8;
        bf16x16 A = join8(*(const bf16x8*)(arow + ka),
                          *(const bf16x8*)(arow + ka + 16));
        bf16x16 B = join8(*(const bf16x8*)(brow + ka),
                          *(const bf16x8*)(brow + ka + 16));
        acc = __builtin_amdgcn_wmma_f32_16x16x32_bf16(
            false, A, false, B, (short)0, acc, false, false);
      }
#pragma unroll
      for (int r = 0; r < 8; ++r)
        gates[(size_t)(m0 + half * 8 + r) * G4 + n] = acc[r];
    }
    __threadfence();
    __syncthreads();

    // phase 2: elementwise gate math; h -> LDS (bf16) + hf (f32)
#pragma unroll 1
    for (int e = tid; e < BATCH * HID; e += 1024) {
      const int b = e >> 9;            // / HID
      const int j = e & (HID - 1);
      const float* gb = gates + (size_t)b * G4;
      float ig = gb[j], fg = gb[HID + j], gg = gb[2 * HID + j], og = gb[3 * HID + j];
      float si = 1.0f / (1.0f + __expf(-ig));
      float sf = 1.0f / (1.0f + __expf(-fg));
      float so = 1.0f / (1.0f + __expf(-og));
      float c  = sf * cbuf[e] + si * tanhf(gg);
      float h  = so * tanhf(c);
      cbuf[e]          = c;
      hf[e]            = h;
      hbf[b * HID + j] = (__bf16)h;
    }
    __threadfence();
    __syncthreads();

    // phase 3: deterministic row-sum, wave w reduces batch b = w
    {
      float sum = 0.0f;
      const float* hr = hf + wv * HID;
#pragma unroll
      for (int j = lane; j < HID; j += 32) sum += hr[j];
#pragma unroll
      for (int m = 16; m >= 1; m >>= 1) sum += __shfl_xor(sum, m, 32);
      if (lane == 0) hsum_out[s * BATCH + wv] = sum;
    }
    __syncthreads();
  }
}

// ---- output: log_softmax of broadcast logits (constant per row) ----------
__global__ void k_out(const float* __restrict__ hsum, float* __restrict__ out) {
  int t  = blockIdx.x * blockDim.x + threadIdx.x;  // exact SEQ*BATCH*TAGS threads
  int sb = t / TAGS;
  float e = 0.5f * (1.0f + __sinf(hsum[sb]));
  out[t]  = e - (e + __logf((float)TAGS));         // == -log(TAGS)
}

extern "C" void kernel_launch(void* const* d_in, const int* in_sizes, int n_in,
                              void* d_out, int out_size, void* d_ws, size_t ws_size,
                              hipStream_t stream) {
  (void)in_sizes; (void)n_in; (void)out_size; (void)ws_size;
  const int*   sent = (const int*)d_in[0];
  // d_in[1] = tagset_size scalar (compile-time TAGS used)
  const float* emb  = (const float*)d_in[2];
  const float* Wih  = (const float*)d_in[3];
  const float* Whh  = (const float*)d_in[4];
  const float* bih  = (const float*)d_in[5];
  const float* bhh  = (const float*)d_in[6];
  float* out = (float*)d_out;

  char* ws = (char*)d_ws;
  size_t off = 0;
  auto alloc = [&](size_t bytes) { void* p = ws + off; off = (off + bytes + 255) & ~(size_t)255; return p; };
  __bf16* xbf    = (__bf16*)alloc((size_t)ROWS * EMB * 2);   // 16 MiB
  __bf16* wih_bf = (__bf16*)alloc((size_t)G4 * EMB * 2);     //  2 MiB
  __bf16* whh_bf = (__bf16*)alloc((size_t)G4 * HID * 2);     //  2 MiB
  float*  bias   = (float*) alloc((size_t)G4 * 4);
  float*  xg     = (float*) alloc((size_t)ROWS * G4 * 4);    // 128 MiB
  float*  gates  = (float*) alloc((size_t)BATCH * G4 * 4);
  float*  cbuf   = (float*) alloc((size_t)BATCH * HID * 4);
  float*  hf     = (float*) alloc((size_t)BATCH * HID * 4);
  float*  hsum   = (float*) alloc((size_t)SEQ * BATCH * 4);

  k_gather_cvt<<<(ROWS * (EMB / 8)) / 256, 256, 0, stream>>>(sent, emb, xbf);
  k_cvt_bf16 <<<(G4 * EMB / 8) / 256, 256, 0, stream>>>(Wih, wih_bf);
  k_cvt_bf16 <<<(G4 * HID / 8) / 256, 256, 0, stream>>>(Whh, whh_bf);
  k_bias     <<<G4 / 256, 256, 0, stream>>>(bih, bhh, bias);

  // 1024 M-tiles * 4 blocks-per-M-row -> 4096 blocks, 8 waves each
  k_gemm_xg  <<<4096, 256, 0, stream>>>(xbf, wih_bf, bias, xg);

  // sequential recurrence: one workgroup (32 wave32s on one WGP)
  k_lstm_seq <<<1, 1024, 0, stream>>>(xg, whh_bf, gates, cbuf, hf, hsum);

  k_out      <<<(SEQ * BATCH * TAGS) / 256, 256, 0, stream>>>(hsum, out);
}
